// MultiStagePointNetEncoder_2963527435034
// MI455X (gfx1250) — matrix operand order repaired and checked by reading, not compiled, not run on hardware
//
#include <hip/hip_runtime.h>
#include <hip/hip_bf16.h>

typedef __attribute__((ext_vector_type(16))) _Float16 v16h;
typedef __attribute__((ext_vector_type(8)))  _Float16 v8h;
typedef __attribute__((ext_vector_type(8)))  float    v8f;

#define B_    32
#define N_    65536
#define KFPS  8
#define NG    (B_ * KFPS)      // 256 groups
#define GRP   512
#define NP    528              // 513 rows padded to 33*16
#define RT_   33               // row tiles of 16
#define H_    128
#define NW_   8                // waves per 256-thread block (wave32)
#define OUTC  128
#define R2_   0.04f
#define BIGF  1e10f
#define EPS_  1e-5f

// ---------------------------------------------------------------------------
// WMMA helpers (gfx1250: V_WMMA_F32_16X16X32_F16, wave32)
// ---------------------------------------------------------------------------
__device__ __forceinline__ v8f wmma_f16(v16h a, v16h b, v8f c) {
    return __builtin_amdgcn_wmma_f32_16x16x32_f16(
        /*neg_a=*/false, a, /*neg_b=*/false, b,
        /*c_mod=*/(short)0, c, /*reuse_a=*/false, /*reuse_b=*/false);
}

// A operand: 16x32 f16 tile from LDS (row-major, stride H_).
// Per lane: M = lane%16, K = hl*16 .. hl*16+15 (contiguous, matching B).
__device__ __forceinline__ v16h load_a_lds(const _Float16* buf, int row, int kc, int hl) {
    const _Float16* p = buf + row * H_ + kc * 32 + hl * 16;
    v8h lo = *(const v8h*)(p);
    v8h hi = *(const v8h*)(p + 8);
    v16h a;
#pragma unroll
    for (int i = 0; i < 8; ++i) { a[i] = lo[i]; a[8 + i] = hi[i]; }
    return a;
}

// B operand: 32x16 f16 tile of W^T. W row-major [128][128] f16 in global.
// lane ln = output column, lanes<16 cover K = kc*32..+15, lanes>=16 K = +16..+31.
__device__ __forceinline__ v16h load_b_w(const _Float16* W, int ct, int kc, int ln, int hl) {
    return *(const v16h*)(W + (ct * 16 + ln) * H_ + kc * 32 + hl * 16);
}

// One full [NP x 128] x [128 x 128]^T pass, row-tile-parallel across 8 waves.
// Modes:
//   accOut == nullptr : out = relu(in @ W^T + bias[col]) written IN PLACE to buf,
//                       optional per-channel running max into ygbits.
//   accOut != nullptr : accOut += in @ W^T   (f16 accumulate, accFirst resets)
__device__ __forceinline__ void gemm_rows(
    _Float16* buf, const _Float16* W, const float* bias,
    unsigned* ygbits, _Float16* accOut, bool accFirst,
    int wv, int ln, int hl)
{
    __builtin_prefetch(W, 0, 3);   // global_prefetch_b8 on the (L2-hot) weights
    for (int rt = wv; rt < RT_; rt += NW_) {
        int arow = rt * 16 + ln;
        v16h a0 = load_a_lds(buf, arow, 0, hl);
        v16h a1 = load_a_lds(buf, arow, 1, hl);
        v16h a2 = load_a_lds(buf, arow, 2, hl);
        v16h a3 = load_a_lds(buf, arow, 3, hl);
#pragma unroll
        for (int ct = 0; ct < 8; ++ct) {
            v8f acc = {};
            acc = wmma_f16(a0, load_b_w(W, ct, 0, ln, hl), acc);
            acc = wmma_f16(a1, load_b_w(W, ct, 1, ln, hl), acc);
            acc = wmma_f16(a2, load_b_w(W, ct, 2, ln, hl), acc);
            acc = wmma_f16(a3, load_b_w(W, ct, 3, ln, hl), acc);
            int col = ct * 16 + ln;
            if (accOut) {
#pragma unroll
                for (int j = 0; j < 8; ++j) {
                    int row = rt * 16 + hl * 8 + j;
                    int idx = row * H_ + col;
                    float prev = accFirst ? 0.0f : (float)accOut[idx];
                    accOut[idx] = (_Float16)(prev + acc[j]);
                }
            } else {
                float bi = bias ? bias[col] : 0.0f;
                float mx = 0.0f;
#pragma unroll
                for (int j = 0; j < 8; ++j) {
                    float v = acc[j] + bi;
                    v = v > 0.0f ? v : 0.0f;
                    mx = fmaxf(mx, v);
                    buf[(rt * 16 + hl * 8 + j) * H_ + col] = (_Float16)v;
                }
                if (ygbits) atomicMax(&ygbits[col], __float_as_uint(mx)); // post-relu >= 0
            }
        }
    }
}

// ---------------------------------------------------------------------------
// Kernel 0: convert weights to f16 (padded / re-sliced) once per launch
// ---------------------------------------------------------------------------
__global__ void wcvt_kernel(const float* conv_in_w, const float* layer_w,
                            const float* glayer_w, const float* conv_out_w,
                            _Float16* Wcin, _Float16* Wlayer,
                            _Float16* Wg1, _Float16* Wout)
{
    int id = blockIdx.x * blockDim.x + threadIdx.x;
    if (id < 4096) {                                   // Wcin[o][k], K padded 3->32
        int o = id >> 5, k = id & 31;
        Wcin[id] = (k < 3) ? (_Float16)conv_in_w[o * 3 + k] : (_Float16)0.0f;
        return;
    }
    id -= 4096;
    if (id < 4 * H_ * H_) { Wlayer[id] = (_Float16)layer_w[id]; return; }
    id -= 4 * H_ * H_;
    if (id < 4 * H_ * H_) {                            // W1 = glayer_w[i][:, :128]
        int i = id >> 14, r = id & 16383, o = r >> 7, k = r & 127;
        Wg1[id] = (_Float16)glayer_w[(i * H_ + o) * (2 * H_) + k];
        return;
    }
    id -= 4 * H_ * H_;
    if (id < 4 * H_ * H_) {                            // Wout_i = conv_out_w[:, i*128:+128]
        int i = id >> 14, r = id & 16383, o = r >> 7, k = r & 127;
        Wout[id] = (_Float16)conv_out_w[o * (4 * H_) + i * H_ + k];
    }
}

// ---------------------------------------------------------------------------
// Kernel 1: farthest point sampling, one block per batch
// ---------------------------------------------------------------------------
__global__ __launch_bounds__(256) void fps_kernel(const float* x, float* dist, float* cxyz)
{
    int b = blockIdx.x, tid = threadIdx.x;
    __shared__ float redV[256];
    __shared__ int   redI[256];
    __shared__ int   centsS[KFPS];
    __shared__ int   curFar;
    if (tid == 0) curFar = 0;
    __syncthreads();
    for (int it = 0; it < KFPS; ++it) {
        int far = curFar;
        if (tid == 0) centsS[it] = far;
        float cx = x[((size_t)b * N_ + far) * 3 + 0];
        float cy = x[((size_t)b * N_ + far) * 3 + 1];
        float cz = x[((size_t)b * N_ + far) * 3 + 2];
        float bestV = -1.0f; int bestI = 0;
        for (int j = tid; j < N_; j += 256) {
            const float* p = x + ((size_t)b * N_ + j) * 3;
            float dx = p[0] - cx, dy = p[1] - cy, dz = p[2] - cz;
            float d = dx * dx + dy * dy + dz * dz;
            float md = (it == 0) ? d : fminf(dist[(size_t)b * N_ + j], d);
            dist[(size_t)b * N_ + j] = md;
            if (md > bestV) { bestV = md; bestI = j; }
        }
        redV[tid] = bestV; redI[tid] = bestI;
        __syncthreads();
        for (int s = 128; s > 0; s >>= 1) {
            if (tid < s && redV[tid + s] > redV[tid]) {
                redV[tid] = redV[tid + s]; redI[tid] = redI[tid + s];
            }
            __syncthreads();
        }
        if (tid == 0) curFar = redI[0];
        __syncthreads();
    }
    if (tid < KFPS) {
        int ci = centsS[tid];
        for (int c = 0; c < 3; ++c)
            cxyz[(b * KFPS + tid) * 3 + c] = x[((size_t)b * N_ + ci) * 3 + c];
    }
}

// ---------------------------------------------------------------------------
// Kernel 2: ball query + grouping (deterministic prefix-scan selection)
// combined[g][0] = centroid abs xyz, [1..512] = rel xyz of first 512 in-radius,
// padded rows zero.
// ---------------------------------------------------------------------------
__global__ __launch_bounds__(256) void ball_group_kernel(const float* x, const float* cxyz,
                                                         float* combined)
{
    int g = blockIdx.x, tid = threadIdx.x;
    int b = g >> 3;
    __shared__ float c3[3];
    __shared__ int   counts[256];
    // zero-fill the full padded group buffer
    for (int i = tid; i < NP * 3; i += 256) combined[(size_t)g * NP * 3 + i] = 0.0f;
    if (tid < 3) c3[tid] = cxyz[g * 3 + tid];
    __syncthreads();
    if (tid < 3) combined[(size_t)g * NP * 3 + tid] = c3[tid];  // row 0 = abs centroid
    const int chunk = N_ / 256;
    const int start = tid * chunk;
    int cnt = 0;
    for (int jj = 0; jj < chunk; ++jj) {
        const float* p = x + ((size_t)b * N_ + start + jj) * 3;
        float dx = p[0] - c3[0], dy = p[1] - c3[1], dz = p[2] - c3[2];
        if (dx * dx + dy * dy + dz * dz <= R2_) ++cnt;
    }
    counts[tid] = cnt;
    __syncthreads();
    if (tid == 0) {  // exclusive scan (serial; 256 elems, trivial)
        int s = 0;
        for (int t = 0; t < 256; ++t) { int c = counts[t]; counts[t] = s; s += c; }
    }
    __syncthreads();
    int pos = counts[tid];
    for (int jj = 0; jj < chunk; ++jj) {
        const float* p = x + ((size_t)b * N_ + start + jj) * 3;
        float dx = p[0] - c3[0], dy = p[1] - c3[1], dz = p[2] - c3[2];
        if (dx * dx + dy * dy + dz * dz <= R2_) {
            if (pos < GRP) {
                float* dst = combined + ((size_t)g * NP + 1 + pos) * 3;
                dst[0] = dx; dst[1] = dy; dst[2] = dz;
            }
            ++pos;
        }
    }
}

// ---------------------------------------------------------------------------
// Kernel 3: main WMMA encoder — one 256-thread block (8 waves) per group.
// LDS: act buffer 528x128 f16 (in-place updates) + f16 accOut 528x128
// ---------------------------------------------------------------------------
__global__ __launch_bounds__(256) void group_encoder_kernel(
    const float* combined,
    const _Float16* Wcin, const _Float16* Wlayer, const _Float16* Wg1, const _Float16* Wout,
    const float* conv_in_b, const float* layer_b,
    const float* glayer_w /*f32, for W2 matvec*/, const float* glayer_b,
    const float* conv_out_b, float* groupOut)
{
    extern __shared__ char smem[];
    _Float16* buf    = (_Float16*)smem;                 // NP*H_ f16
    _Float16* accOut = buf + NP * H_;                   // NP*H_ f16
    unsigned* ygbits = (unsigned*)(accOut + NP * H_);   // H_ u32
    float*    gbias  = (float*)(ygbits + H_);           // H_ f32

    int g   = blockIdx.x;
    int tid = threadIdx.x;
    int ln  = tid & 15;
    int hl  = (tid >> 4) & 1;
    int wv  = tid >> 5;

    // ---- Stage IN: buf = relu(combined @ conv_in_w^T + b), K padded 3->32 ----
    for (int rt = wv; rt < RT_; rt += NW_) {
        v16h a = {};
        int row = rt * 16 + ln;
        const float* src = combined + ((size_t)g * NP + row) * 3;
        if (hl == 0) {  // lanes 0-15 carry K=0..7 in the first 8 slots
            a[0] = (_Float16)src[0];
            a[1] = (_Float16)src[1];
            a[2] = (_Float16)src[2];
        }
#pragma unroll
        for (int ct = 0; ct < 8; ++ct) {
            v16h bmat = *(const v16h*)(Wcin + (ct * 16 + ln) * 32 + hl * 16);
            v8f acc = {};
            acc = wmma_f16(a, bmat, acc);
            int col = ct * 16 + ln;
            float bi = conv_in_b[col];
#pragma unroll
            for (int j = 0; j < 8; ++j) {
                float v = acc[j] + bi;
                v = v > 0.0f ? v : 0.0f;
                buf[(rt * 16 + hl * 8 + j) * H_ + col] = (_Float16)v;
            }
        }
    }
    __syncthreads();

    // ---- 4 layers ----
    for (int i = 0; i < 4; ++i) {
        if (tid < H_) ygbits[tid] = 0u;
        __syncthreads();
        // t = relu(y @ layer_w^T + b), track per-channel max (yg)
        gemm_rows(buf, Wlayer + i * H_ * H_, layer_b + i * H_, ygbits, nullptr, false, wv, ln, hl);
        __syncthreads();
        // gbias[o] = glayer_b[o] + sum_k yg[k] * W2[o][k]   (W2 = glayer_w[i][:,128:])
        if (tid < H_) {
            float s = glayer_b[i * H_ + tid];
            const float* w2 = glayer_w + ((size_t)(i * H_) + tid) * (2 * H_) + H_;
            for (int k = 0; k < H_; ++k) s += __uint_as_float(ygbits[k]) * w2[k];
            gbias[tid] = s;
        }
        __syncthreads();
        // y' = relu(t @ W1^T + gbias), in place
        gemm_rows(buf, Wg1 + i * H_ * H_, gbias, nullptr, nullptr, false, wv, ln, hl);
        __syncthreads();
        // accOut += y' @ Wout_i^T
        gemm_rows(buf, Wout + i * H_ * H_, nullptr, nullptr, accOut, (i == 0), wv, ln, hl);
        __syncthreads();
    }

    // ---- per-group channel max over the 513 real rows ----
    if (tid < H_) {
        int o = tid;
        float bo = conv_out_b[o];
        float m = -3.0e38f;
        for (int r = 0; r < GRP + 1; ++r)
            m = fmaxf(m, (float)accOut[r * H_ + o] + bo);
        groupOut[g * OUTC + o] = m;
    }
}

// ---------------------------------------------------------------------------
// Kernel 4: y_global = max over the 8 groups of each batch
// ---------------------------------------------------------------------------
__global__ void reduce_y_kernel(const float* groupOut, float* out)
{
    int b = blockIdx.x, o = threadIdx.x;
    float m = -3.0e38f;
    for (int k = 0; k < KFPS; ++k)
        m = fmaxf(m, groupOut[(b * KFPS + k) * OUTC + o]);
    out[b * OUTC + o] = m;
}

// ---------------------------------------------------------------------------
// Kernel 5/6: centroid MLP branch (tiny; 256 rows total)
// ---------------------------------------------------------------------------
__device__ __forceinline__ void ln_act(float* h, int n, const float* g, const float* be, bool relu)
{
    float m = 0.0f;
    for (int i = 0; i < n; ++i) m += h[i];
    m /= n;
    float v = 0.0f;
    for (int i = 0; i < n; ++i) { float d = h[i] - m; v += d * d; }
    v /= n;
    float inv = rsqrtf(v + EPS_);
    for (int i = 0; i < n; ++i) {
        float y = (h[i] - m) * inv * g[i] + be[i];
        h[i] = (relu && y < 0.0f) ? 0.0f : y;
    }
}

__global__ void cent_mlp_kernel(const float* cxyz,
                                const float* w0, const float* b0, const float* g0, const float* e0,
                                const float* w1, const float* b1, const float* g1, const float* e1,
                                const float* w2, const float* b2, const float* g2, const float* e2,
                                float* cfeat)
{
    int t = blockIdx.x * blockDim.x + threadIdx.x;
    if (t >= NG) return;
    float c0 = cxyz[t * 3 + 0], c1 = cxyz[t * 3 + 1], c2v = cxyz[t * 3 + 2];
    float h0[64];
    for (int o = 0; o < 64; ++o)
        h0[o] = b0[o] + w0[o * 3] * c0 + w0[o * 3 + 1] * c1 + w0[o * 3 + 2] * c2v;
    ln_act(h0, 64, g0, e0, true);
    float h1[128];
    for (int o = 0; o < 128; ++o) {
        float s = b1[o];
        for (int j = 0; j < 64; ++j) s += w1[o * 64 + j] * h0[j];
        h1[o] = s;
    }
    ln_act(h1, 128, g1, e1, true);
    float h2[256];
    for (int o = 0; o < 256; ++o) {
        float s = b2[o];
        for (int j = 0; j < 128; ++j) s += w2[o * 128 + j] * h1[j];
        h2[o] = s;
    }
    ln_act(h2, 256, g2, e2, true);
    for (int o = 0; o < 256; ++o) cfeat[t * 256 + o] = h2[o];
}

__global__ __launch_bounds__(128) void cent_final_kernel(const float* cfeat,
                                                         const float* fp_w, const float* fp_b,
                                                         const float* fg, const float* fe,
                                                         float* out)
{
    int b = blockIdx.x, tid = threadIdx.x;
    __shared__ float cmax[256];
    __shared__ float red[128];
    for (int o = tid; o < 256; o += 128) {
        float m = -3.0e38f;
        for (int k = 0; k < KFPS; ++k) m = fmaxf(m, cfeat[(b * KFPS + k) * 256 + o]);
        cmax[o] = m;
    }
    __syncthreads();
    float s = fp_b[tid];
    for (int j = 0; j < 256; ++j) s += cmax[j] * fp_w[tid * 256 + j];
    red[tid] = s;
    __syncthreads();
    for (int st = 64; st > 0; st >>= 1) {
        if (tid < st) red[tid] += red[tid + st];
        __syncthreads();
    }
    float mean = red[0] / 128.0f;
    __syncthreads();
    float d = s - mean;
    red[tid] = d * d;
    __syncthreads();
    for (int st = 64; st > 0; st >>= 1) {
        if (tid < st) red[tid] += red[tid + st];
        __syncthreads();
    }
    float var = red[0] / 128.0f;
    out[b * OUTC + tid] = d * rsqrtf(var + EPS_) * fg[tid] + fe[tid];
}

// ---------------------------------------------------------------------------
extern "C" void kernel_launch(void* const* d_in, const int* in_sizes, int n_in,
                              void* d_out, int out_size, void* d_ws, size_t ws_size,
                              hipStream_t stream) {
    (void)in_sizes; (void)n_in; (void)out_size; (void)ws_size;
    const float* x          = (const float*)d_in[0];
    const float* conv_in_w  = (const float*)d_in[1];
    const float* conv_in_b  = (const float*)d_in[2];
    const float* layer_w    = (const float*)d_in[3];
    const float* layer_b    = (const float*)d_in[4];
    const float* glayer_w   = (const float*)d_in[5];
    const float* glayer_b   = (const float*)d_in[6];
    const float* conv_out_w = (const float*)d_in[7];
    const float* conv_out_b = (const float*)d_in[8];
    float* out = (float*)d_out;

    // workspace carve-up (256B aligned)
    char* ws = (char*)d_ws;
    size_t off = 0;
    auto take = [&](size_t bytes) { void* p = ws + off; off += (bytes + 255) & ~size_t(255); return p; };
    float*     dist     = (float*)take((size_t)B_ * N_ * 4);          // 8 MB
    float*     cxyz     = (float*)take((size_t)NG * 3 * 4);
    float*     combined = (float*)take((size_t)NG * NP * 3 * 4);      // 1.6 MB
    _Float16*  Wcin     = (_Float16*)take((size_t)H_ * 32 * 2);
    _Float16*  Wlayer   = (_Float16*)take((size_t)4 * H_ * H_ * 2);
    _Float16*  Wg1      = (_Float16*)take((size_t)4 * H_ * H_ * 2);
    _Float16*  Wout     = (_Float16*)take((size_t)4 * H_ * H_ * 2);
    float*     groupOut = (float*)take((size_t)NG * OUTC * 4);
    float*     cfeat    = (float*)take((size_t)NG * 256 * 4);

    // 0. weights -> f16 (padded / re-sliced)
    {
        int total = 4096 + 3 * 4 * H_ * H_;
        wcvt_kernel<<<(total + 255) / 256, 256, 0, stream>>>(
            conv_in_w, layer_w, glayer_w, conv_out_w, Wcin, Wlayer, Wg1, Wout);
    }
    // 1. FPS
    fps_kernel<<<B_, 256, 0, stream>>>(x, dist, cxyz);
    // 2. ball query + grouping
    ball_group_kernel<<<NG, 256, 0, stream>>>(x, cxyz, combined);
    // 3. WMMA encoder, one workgroup per group; dynamic LDS 2*(528*128) f16 + 1 KB
    {
        size_t smemBytes = (size_t)2 * NP * H_ * sizeof(_Float16) + H_ * 4 + H_ * 4;
        group_encoder_kernel<<<NG, 256, smemBytes, stream>>>(
            combined, Wcin, Wlayer, Wg1, Wout,
            conv_in_b, layer_b, glayer_w, glayer_b, conv_out_b, groupOut);
    }
    // 4. y_global
    reduce_y_kernel<<<B_, OUTC, 0, stream>>>(groupOut, out);
    // 5/6. centroid branch
    cent_mlp_kernel<<<1, 256, 0, stream>>>(
        cxyz,
        (const float*)d_in[9],  (const float*)d_in[10], (const float*)d_in[11], (const float*)d_in[12],
        (const float*)d_in[13], (const float*)d_in[14], (const float*)d_in[15], (const float*)d_in[16],
        (const float*)d_in[17], (const float*)d_in[18], (const float*)d_in[19], (const float*)d_in[20],
        cfeat);
    cent_final_kernel<<<B_, 128, 0, stream>>>(
        cfeat, (const float*)d_in[21], (const float*)d_in[22],
        (const float*)d_in[23], (const float*)d_in[24], out + B_ * OUTC);
}
